// DoReFaConv2d_75170517614967
// MI455X (gfx1250) — 8-bit matrix operand order repaired and checked by reading, compiled, not run on hardware
//
#include <hip/hip_runtime.h>


typedef int   v16i __attribute__((ext_vector_type(16)));
typedef float v8f  __attribute__((ext_vector_type(8)));

#define NB    32
#define CIN   256
#define HH    56
#define WW    56
#define COUT  256
#define HP    58
#define WP    58
#define PIX   (HH*WW)
#define KTILES 18
#define COLS  64
#define LDST  144
#define XQ_BYTES ((long)NB*HP*WP*CIN)
#define WQ_BYTES ((long)COUT*CIN*9)

#define BOFF(it) ((((((it) >> 1) / 3) * WP + (((it) >> 1) % 3)) * CIN) + ((it) & 1) * 128)

union Frag { v16i v; int4 q[4]; };

__global__ __launch_bounds__(256) void zero_xq(int4* p, int n16) {
    int i = blockIdx.x * 256 + threadIdx.x;
    if (i < n16) p[i] = make_int4(0, 0, 0, 0);
}

__global__ __launch_bounds__(256) void quant_x(const float* __restrict__ x,
                                               unsigned char* __restrict__ xq) {
    int tid = blockIdx.x * 256 + threadIdx.x;
    int p   = tid % PIX;
    int cg  = (tid / PIX) & 15;
    int n   = tid / (PIX * 16);
    int h = p / WW, w = p % WW;

    unsigned int words[4];
#pragma unroll
    for (int j = 0; j < 4; ++j) {
        unsigned int wd = 0;
#pragma unroll
        for (int k = 0; k < 4; ++k) {
            int c = cg * 16 + j * 4 + k;
            float v = __builtin_nontemporal_load(&x[((long)(n * CIN + c)) * PIX + p]);
            unsigned int b = (v > 0.f) ? 0x38u : ((v < 0.f) ? 0xB8u : 0x00u);
            wd |= b << (8 * k);
        }
        words[j] = wd;
    }
    long dst = (((long)(n * HP + h + 1) * WP) + (w + 1)) * CIN + cg * 16;
    *(int4*)(xq + dst) = make_int4((int)words[0], (int)words[1],
                                   (int)words[2], (int)words[3]);
}

__global__ __launch_bounds__(256) void quant_w(const float* __restrict__ w,
                                               unsigned char* __restrict__ wq) {
    int id   = blockIdx.x * 256 + threadIdx.x;
    int tile = id >> 11;
    int off  = id & 2047;
    int lane = off >> 6;
    int b    = off & 63;
    int mo   = tile / KTILES;
    int tc   = tile % KTILES;
    int tap  = tc >> 1;
    int kc   = tc & 1;
    int r = tap / 3, s = tap % 3;
    int t  = b >> 3;
    int kk = ((b >> 4) << 5) + ((lane >> 4) << 4) + (b & 15);
    int c  = (kc << 7) + kk;
    int m  = (mo << 4) + (lane & 15);
    float v = w[(((long)(m * CIN + c)) * 3 + r) * 3 + s];
    wq[id] = (v > 0.f) ? 0x38u : ((v < 0.f) ? 0xB8u : 0x00u);
}

__global__ __launch_bounds__(256) void wscale(const float* __restrict__ w,
                                              float* __restrict__ scale) {
    int m = threadIdx.x;
    const float* row = w + (long)m * (CIN * 9);
    float s = 0.f;
    for (int i = 0; i < CIN * 9; ++i) s += fabsf(row[i]);
    scale[m] = s * (1.0f / (CIN * 9));
}

__device__ __forceinline__ void load_a(const unsigned char* __restrict__ abase,
                                       int aoff, Frag& A) {
    const int4* ap = (const int4*)(abase + aoff);
    A.q[0] = ap[0]; A.q[1] = ap[1]; A.q[2] = ap[2]; A.q[3] = ap[3];
}

__global__ __launch_bounds__(256) void dorefa_gemm(
        const unsigned char* __restrict__ xq,
        const unsigned char* __restrict__ wq,
        const float* __restrict__ scale,
        float* __restrict__ out) {
    __shared__ unsigned char smem[2][COLS * LDST];

    const int tid  = threadIdx.x;
    const int lane = tid & 31;
    const int wv   = tid >> 5;
    const int bid  = blockIdx.x;
    const int n    = bid / 49;
    const int pbase = (bid % 49) * COLS;

    const int col0 = tid >> 3, chk = (tid & 7) * 16;
    const int col1 = col0 + 32;
    const int p0 = pbase + col0, p1 = pbase + col1;
    const unsigned char* g0 =
        xq + ((long)(n * HP + p0 / WW) * WP + p0 % WW) * CIN + chk;
    const unsigned char* g1 =
        xq + ((long)(n * HP + p1 / WW) * WP + p1 % WW) * CIN + chk;
    unsigned char* l0 = &smem[0][col0 * LDST + chk];
    unsigned char* l1 = &smem[0][col1 * LDST + chk];

    *(int4*)l0 = *(const int4*)(g0 + BOFF(0));
    *(int4*)l1 = *(const int4*)(g1 + BOFF(0));
    __syncthreads();

    const unsigned char* a0base = wq + (long)((wv * 2 + 0) * KTILES) * 2048 + lane * 64;
    const unsigned char* a1base = wq + (long)((wv * 2 + 1) * KTILES) * 2048 + lane * 64;
    const int khalf = (lane >> 4) << 4;
    const unsigned char* bl = &smem[0][(lane & 15) * LDST + khalf];

    v8f acc[2][4];
#pragma unroll
    for (int mt = 0; mt < 2; ++mt)
#pragma unroll
        for (int s = 0; s < 4; ++s) acc[mt][s] = (v8f){};

#pragma unroll
    for (int it = 0; it < KTILES; ++it) {
        const int cur = it & 1;
        const int nxt = cur ^ 1;

        int4 s0, s1;
        if (it < KTILES - 1) {
            s0 = *(const int4*)(g0 + BOFF(it + 1));
            s1 = *(const int4*)(g1 + BOFF(it + 1));
        }

        Frag A0, A1;
        load_a(a0base, it * 2048, A0);
        load_a(a1base, it * 2048, A1);

#pragma unroll
        for (int s = 0; s < 4; ++s) {
            Frag B;
            const unsigned char* bp = bl + cur * (COLS * LDST) + s * 16 * LDST;
            B.q[0] = *(const int4*)(bp);
            B.q[1] = *(const int4*)(bp + 32);
            B.q[2] = *(const int4*)(bp + 64);
            B.q[3] = *(const int4*)(bp + 96);
            acc[0][s] = __builtin_amdgcn_wmma_f32_16x16x128_fp8_fp8(
                            A0.v, B.v, (short)0, acc[0][s], false, false);
            acc[1][s] = __builtin_amdgcn_wmma_f32_16x16x128_fp8_fp8(
                            A1.v, B.v, (short)0, acc[1][s], false, false);
        }

        if (it < KTILES - 1) {
            *(int4*)(l0 + nxt * (COLS * LDST)) = s0;
            *(int4*)(l1 + nxt * (COLS * LDST)) = s1;
            __syncthreads();
        }
    }

    const int colb = lane & 15;
#pragma unroll
    for (int mt = 0; mt < 2; ++mt) {
        const int moff = (wv * 2 + mt) * 16 + ((lane >> 4) << 3);
        float sc[8];
#pragma unroll
        for (int d = 0; d < 8; ++d) sc[d] = scale[moff + d];
#pragma unroll
        for (int d = 0; d < 8; ++d) {
            const long orow = ((long)n * COUT + (moff + d)) * PIX + pbase + colb;
#pragma unroll
            for (int s = 0; s < 4; ++s)
                __builtin_nontemporal_store(sc[d] * acc[mt][s][d],
                                            &out[orow + s * 16]);
        }
    }
}

extern "C" void kernel_launch(void* const* d_in, const int* in_sizes, int n_in,
                              void* d_out, int out_size, void* d_ws, size_t ws_size,
                              hipStream_t stream) {
    const float* x = (const float*)d_in[0];
    const float* w = (const float*)d_in[1];
    unsigned char* xq = (unsigned char*)d_ws;
    unsigned char* wq = xq + XQ_BYTES;
    float* scale = (float*)(wq + WQ_BYTES);
    float* out = (float*)d_out;

    int n16 = (int)(XQ_BYTES / 16);
    zero_xq<<<(n16 + 255) / 256, 256, 0, stream>>>((int4*)xq, n16);
    quant_x<<<(NB * 16 * PIX) / 256, 256, 0, stream>>>(x, xq);
    quant_w<<<(int)(WQ_BYTES / 256), 256, 0, stream>>>(w, wq);
    wscale<<<1, 256, 0, stream>>>(w, scale);
    dorefa_gemm<<<NB * 49, 256, 0, stream>>>(xq, wq, scale, out);
}
